// EdgeFeatureNet_19378892439932
// MI455X (gfx1250) — matrix operand order repaired and checked by reading, not compiled, not run on hardware
//
#include <hip/hip_runtime.h>
#include <hip/hip_bf16.h>

typedef __attribute__((ext_vector_type(16))) _Float16 v16h;
typedef __attribute__((ext_vector_type(4)))  _Float16 v4h;
typedef __attribute__((ext_vector_type(8)))  float    v8f;

#define NRES  768
#define CS    256
#define FEATD 64
#define CP    128
#define EPSLN 1e-5f
#define MAXLEN 2056.0f
#define MTILE 64                     // pairs per block
#define JBLK  (NRES / MTILE)         // 12 blocks per i-row

// ---------------------------------------------------------------------------
// K1: p[i][f] = s[i] @ W_sp + b_sp      (768 x 64)
// ---------------------------------------------------------------------------
__global__ void k_node_proj(const float* __restrict__ s,
                            const float* __restrict__ Wsp,
                            const float* __restrict__ bsp,
                            float* __restrict__ p) {
  int i = blockIdx.x;
  int f = threadIdx.x;               // 64 threads
  const float* srow = s + i * CS;
  float acc = bsp[f];
  for (int k = 0; k < CS; ++k) acc += srow[k] * Wsp[k * FEATD + f];
  p[i * FEATD + f] = acc;
}

// ---------------------------------------------------------------------------
// K2: per-node layer-1 contribution tables
//   T[i][c] = sum_f p[i][f]*W1[rowOff+f][c] + dm[i]*W1[dmRow][c] + 0.5*b1[c]
// ---------------------------------------------------------------------------
__global__ void k_node_table(const float* __restrict__ p,
                             const float* __restrict__ W1,
                             const float* __restrict__ b1,
                             const float* __restrict__ dm,
                             int rowOff, int dmRow,
                             float* __restrict__ T) {
  int i = blockIdx.x;
  int c = threadIdx.x;               // 128 threads
  float acc = 0.5f * b1[c] + dm[i] * W1[dmRow * CP + c];
  const float* prow = p + i * FEATD;
  for (int f = 0; f < FEATD; ++f) acc += prow[f] * W1[(rowOff + f) * CP + c];
  T[i * CP + c] = acc;
}

// ---------------------------------------------------------------------------
// K3: relpos table R[d][c] = (emb(d) @ W_rp + b_rp) @ W1[128:192]
// ---------------------------------------------------------------------------
__global__ void k_relpos(const float* __restrict__ Wrp,
                         const float* __restrict__ brp,
                         const float* __restrict__ W1,
                         float* __restrict__ R) {
  __shared__ float e[FEATD];
  __shared__ float rp[FEATD];
  int dIdx = blockIdx.x;
  float d = (float)(dIdx - (NRES - 1));
  int tid = threadIdx.x;             // 128 threads
  if (tid < 32) {
    float ang = d * 3.14159265358979323846f /
                powf(MAXLEN, (2.0f * (float)tid) / (float)FEATD);
    e[tid]      = sinf(ang);
    e[tid + 32] = cosf(ang);
  }
  __syncthreads();
  if (tid < FEATD) {
    float a = brp[tid];
    for (int k = 0; k < FEATD; ++k) a += e[k] * Wrp[k * FEATD + tid];
    rp[tid] = a;
  }
  __syncthreads();
  float acc = 0.f;
  for (int f = 0; f < FEATD; ++f) acc += rp[f] * W1[(128 + f) * CP + tid];
  R[dIdx * CP + tid] = acc;
}

// ---------------------------------------------------------------------------
// K4: swizzle 128x128 fp32 weight into f16 WMMA B-fragment layout.
//   lane n = l&15, hi = l>>4, element e -> K = 32*s + 16*(e>>3) + 8*hi + (e&7)
// ---------------------------------------------------------------------------
__global__ void k_wswz(const float* __restrict__ W, _Float16* __restrict__ Wf) {
  int idx  = blockIdx.x * blockDim.x + threadIdx.x;   // 16384 slots
  int e    = idx & 15;
  int lane = (idx >> 4) & 31;
  int s    = (idx >> 9) & 3;
  int t    = (idx >> 11) & 7;
  int n    = t * 16 + (lane & 15);
  int hi   = lane >> 4;
  int kg   = s * 32 + ((e & 8) << 1) + hi * 8 + (e & 7);
  Wf[idx]  = (_Float16)W[kg * CP + n];
}

// ---------------------------------------------------------------------------
// A-fragment swizzle helpers (ISA 7.12.2, 16-bit A 16x32).
// Aligned 4-channel groups map to 4 consecutive elements of one lane.
// ---------------------------------------------------------------------------
__device__ __forceinline__ int a_slot4(int c, int ml) {   // c multiple of 4
  int s  = c >> 5;
  int kl = c & 31;
  int hi = (kl >> 3) & 1;
  int kk = kl - hi * 8;
  int vr = (kk < 8) ? (kk >> 1) : (4 + ((kk - 16) >> 1));
  return (s * 32 + hi * 16 + ml) * 16 + vr * 2;
}
__device__ __forceinline__ int a_slot1(int c, int ml) {
  int s  = c >> 5;
  int kl = c & 31;
  int hi = (kl >> 3) & 1;
  int kk = kl - hi * 8;
  int vr = (kk < 8) ? (kk >> 1) : (4 + ((kk - 16) >> 1));
  return (s * 32 + hi * 16 + ml) * 16 + vr * 2 + (kl & 1);
}

// ---------------------------------------------------------------------------
// Main fused kernel: 64 pairs/block (4 m-tiles), 8 wave32s, wave w = col tile.
//   layer1: table-composed, ReLU -> f16 LDS (A-swizzle, packed b64 stores)
//   layer2: 16x v_wmma_f32_16x16x32_f16, +b2, ReLU -> LDS (A-swizzle)
//   layer3: 16x v_wmma_f32_16x16x32_f16, +b3 -> LDS fp32
//   layernorm (width-4 shfl_xor) + mask + float4 stores
// ---------------------------------------------------------------------------
__global__ void __launch_bounds__(256) k_edge_main(
    const float* __restrict__ t3,  const float* __restrict__ sc3,
    const float* __restrict__ pmask,
    const float* __restrict__ W1,
    const float* __restrict__ A,   const float* __restrict__ Bt,
    const float* __restrict__ R,
    const _Float16* __restrict__ W2f, const _Float16* __restrict__ W3f,
    const float* __restrict__ b2,  const float* __restrict__ b3,
    const float* __restrict__ lng, const float* __restrict__ lnb,
    float* __restrict__ out) {
  __shared__ __align__(16) _Float16 h1[MTILE / 16 * 2048];   // 16 KB
  __shared__ __align__(16) _Float16 h2[MTILE / 16 * 2048];   // 16 KB
  __shared__ __align__(16) float    h3[MTILE * CP];          // 32 KB
  __shared__ __align__(16) float    shA[CP];                 // A[i] row

  int tid  = threadIdx.x;
  int lane = tid & 31;
  int w    = tid >> 5;                  // wave id == output column tile
  int i    = blockIdx.x / JBLK;         // fixed row index for this block
  int j0   = (blockIdx.x - i * JBLK) * MTILE;

  // Per-wave weight fragments, resident in VGPRs, reused across 4 m-tiles.
  v16h wb2[4], wb3[4];
#pragma unroll
  for (int s = 0; s < 4; ++s) {
    wb2[s] = *(const v16h*)(W2f + ((w * 4 + s) * 32 + lane) * 16);
    wb3[s] = *(const v16h*)(W3f + ((w * 4 + s) * 32 + lane) * 16);
  }

  // Stage A[i] row once (kills 64x redundant global reads).
  if (tid < CP) shA[tid] = A[i * CP + tid];
  __syncthreads();

  // -------- layer 1 : shA + Bt[j] + R[i-j] + W1dist[bin] lookups --------
  {
    int m  = tid >> 2;                 // pair row within block (0..63)
    int q  = tid & 3;                  // 32-channel group
    int j  = j0 + m;
    int mt = m >> 4, ml = m & 15;

    float dx = t3[i * 3 + 0] - t3[j * 3 + 0];
    float dy = t3[i * 3 + 1] - t3[j * 3 + 1];
    float dz = t3[i * 3 + 2] - t3[j * 3 + 2];
    float dd = sqrtf(dx * dx + dy * dy + dz * dz);
    float ex = sc3[i * 3 + 0] - sc3[j * 3 + 0];
    float ey = sc3[i * 3 + 1] - sc3[j * 3 + 1];
    float ez = sc3[i * 3 + 2] - sc3[j * 3 + 2];
    float de = sqrtf(ex * ex + ey * ey + ez * ez);

    const float step = (20.0f - 0.001f) / 21.0f;
    int bt = (dd > 0.001f) ? (int)((dd - 0.001f) / step) : -1;
    if (bt > 21) bt = 21;
    int bs = (de > 0.001f) ? (int)((de - 0.001f) / step) : -1;
    if (bs > 21) bs = 21;

    const float4* vb = (const float4*)(Bt + j * CP);
    const float4* vr = (const float4*)(R + (i - j + (NRES - 1)) * CP);
    const float4* vt = (bt >= 0) ? (const float4*)(W1 + (192 + bt) * CP) : nullptr;
    const float4* ve = (bs >= 0) ? (const float4*)(W1 + (214 + bs) * CP) : nullptr;
    const float4* va = (const float4*)shA;

    int c0 = q * 32;
#pragma unroll
    for (int g = 0; g < 8; ++g) {
      int c  = c0 + g * 4;
      int g4 = c >> 2;
      float4 fa = va[g4];
      float4 fb = vb[g4];
      float4 fr = vr[g4];
      float x0 = fa.x + fb.x + fr.x;
      float x1 = fa.y + fb.y + fr.y;
      float x2 = fa.z + fb.z + fr.z;
      float x3 = fa.w + fb.w + fr.w;
      if (vt) { float4 f = vt[g4]; x0 += f.x; x1 += f.y; x2 += f.z; x3 += f.w; }
      if (ve) { float4 f = ve[g4]; x0 += f.x; x1 += f.y; x2 += f.z; x3 += f.w; }
      v4h pk;
      pk.x = (_Float16)fmaxf(x0, 0.f);
      pk.y = (_Float16)fmaxf(x1, 0.f);
      pk.z = (_Float16)fmaxf(x2, 0.f);
      pk.w = (_Float16)fmaxf(x3, 0.f);
      *(v4h*)&h1[mt * 2048 + a_slot4(c, ml)] = pk;   // ds_store_b64
    }
  }
  __syncthreads();

  // -------- layer 2 : 4 m-tiles x 4 k-steps of WMMA --------
  {
    int n = lane & 15, hil = lane >> 4;
    int c = w * 16 + n;
    float bias = b2[c];
#pragma unroll
    for (int mt = 0; mt < 4; ++mt) {
      v8f acc = {};
#pragma unroll
      for (int s = 0; s < 4; ++s) {
        v16h a = *(const v16h*)(&h1[mt * 2048 + (s * 32 + lane) * 16]);
        acc = __builtin_amdgcn_wmma_f32_16x16x32_f16(
            false, a, false, wb2[s], (short)0, acc, false, false);
      }
#pragma unroll
      for (int v = 0; v < 8; ++v) {
        float x = fmaxf(acc[v] + bias, 0.f);
        int ml = v + hil * 8;
        h2[mt * 2048 + a_slot1(c, ml)] = (_Float16)x;
      }
    }
  }
  __syncthreads();

  // -------- layer 3 : 4 m-tiles x 4 k-steps of WMMA --------
  {
    int n = lane & 15, hil = lane >> 4;
    float bias = b3[w * 16 + n];
#pragma unroll
    for (int mt = 0; mt < 4; ++mt) {
      v8f acc = {};
#pragma unroll
      for (int s = 0; s < 4; ++s) {
        v16h a = *(const v16h*)(&h2[mt * 2048 + (s * 32 + lane) * 16]);
        acc = __builtin_amdgcn_wmma_f32_16x16x32_f16(
            false, a, false, wb3[s], (short)0, acc, false, false);
      }
#pragma unroll
      for (int v = 0; v < 8; ++v)
        h3[(mt * 16 + v + hil * 8) * CP + w * 16 + n] = acc[v] + bias;
    }
  }
  __syncthreads();

  // -------- layernorm + mask + store --------
  {
    int r = tid >> 2;                  // pair row (0..63)
    int q = tid & 3;                   // 32 channels each
    const float4* row4 = (const float4*)(h3 + r * CP);
    float4 xs[8];
    float sum = 0.f, sq = 0.f;
#pragma unroll
    for (int g = 0; g < 8; ++g) {
      float4 x = row4[q * 8 + g];
      xs[g] = x;
      sum += x.x + x.y + x.z + x.w;
      sq  += x.x * x.x + x.y * x.y + x.z * x.z + x.w * x.w;
    }
    // 4 threads per row sit in consecutive lanes: width-4 xor reduction
#pragma unroll
    for (int mskb = 1; mskb < 4; mskb <<= 1) {
      sum += __shfl_xor(sum, mskb, 4);
      sq  += __shfl_xor(sq,  mskb, 4);
    }
    float mu  = sum * (1.0f / CP);
    float var = sq  * (1.0f / CP) - mu * mu;
    float inv = rsqrtf(var + EPSLN);

    size_t pIdx = (size_t)i * NRES + (j0 + r);
    float msk = pmask[pIdx];
    float4* op = (float4*)(out + pIdx * CP + q * 32);
    const float4* g4 = (const float4*)(lng + q * 32);
    const float4* bb4 = (const float4*)(lnb + q * 32);
#pragma unroll
    for (int g = 0; g < 8; ++g) {
      float4 gg = g4[g], bbv = bb4[g], x = xs[g];
      float4 o;
      o.x = ((x.x - mu) * inv * gg.x + bbv.x) * msk;
      o.y = ((x.y - mu) * inv * gg.y + bbv.y) * msk;
      o.z = ((x.z - mu) * inv * gg.z + bbv.z) * msk;
      o.w = ((x.w - mu) * inv * gg.w + bbv.w) * msk;
      op[g] = o;
    }
  }
}

// ---------------------------------------------------------------------------
extern "C" void kernel_launch(void* const* d_in, const int* in_sizes, int n_in,
                              void* d_out, int out_size, void* d_ws, size_t ws_size,
                              hipStream_t stream) {
  const float* s   = (const float*)d_in[0];
  const float* t   = (const float*)d_in[1];
  const float* sct = (const float*)d_in[2];
  const float* pm  = (const float*)d_in[3];
  const float* dm  = (const float*)d_in[4];
  const float* Wsp = (const float*)d_in[5];
  const float* bsp = (const float*)d_in[6];
  const float* Wrp = (const float*)d_in[7];
  const float* brp = (const float*)d_in[8];
  const float* W1  = (const float*)d_in[9];
  const float* b1  = (const float*)d_in[10];
  const float* W2  = (const float*)d_in[11];
  const float* b2  = (const float*)d_in[12];
  const float* W3  = (const float*)d_in[13];
  const float* b3  = (const float*)d_in[14];
  const float* lng = (const float*)d_in[15];
  const float* lnb = (const float*)d_in[16];
  float* out = (float*)d_out;

  // workspace layout (fp32 tables, then f16 swizzled weights)
  float* p  = (float*)d_ws;                 // 768*64
  float* A  = p  + NRES * FEATD;            // 768*128
  float* Bt = A  + NRES * CP;               // 768*128
  float* R  = Bt + NRES * CP;               // 1535*128
  _Float16* W2f = (_Float16*)(R + (2 * NRES - 1) * CP);  // 128*128 f16
  _Float16* W3f = W2f + CP * CP;

  k_node_proj<<<NRES, FEATD, 0, stream>>>(s, Wsp, bsp, p);
  k_node_table<<<NRES, CP, 0, stream>>>(p, W1, b1, dm, 0,   236, A);
  k_node_table<<<NRES, CP, 0, stream>>>(p, W1, b1, dm, 64,  237, Bt);
  k_relpos<<<2 * NRES - 1, CP, 0, stream>>>(Wrp, brp, W1, R);
  k_wswz<<<64, 256, 0, stream>>>(W2, W2f);
  k_wswz<<<64, 256, 0, stream>>>(W3, W3f);

  int nblocks = (NRES * NRES) / MTILE;      // 9216 blocks, 64 pairs each
  k_edge_main<<<nblocks, 256, 0, stream>>>(t, sct, pm, W1, A, Bt, R,
                                           W2f, W3f, b2, b3, lng, lnb, out);
}
